// DiagonalBlock_85581518340398
// MI455X (gfx1250) — matrix operand order repaired and checked by reading, not compiled, not run on hardware
//
#include <hip/hip_runtime.h>
#include <hip/hip_bf16.h>
#include <math.h>

typedef __attribute__((ext_vector_type(16))) __bf16 bf16x16;
typedef __attribute__((ext_vector_type(8)))  __bf16 bf16x8;
typedef __attribute__((ext_vector_type(8)))  float  f32x8;
typedef __attribute__((ext_vector_type(4)))  int    i32x4;

#define BK 64   // k-depth of an LDS tile (2 WMMA k-steps)

#define AS1 __attribute__((address_space(1)))
#define AS3 __attribute__((address_space(3)))

// gfx1250 async global->LDS (ASYNCcnt path); fallback to reg-staged loads.
#if defined(__has_builtin)
#  if __has_builtin(__builtin_amdgcn_global_load_async_to_lds_b128) && \
      __has_builtin(__builtin_amdgcn_s_wait_asynccnt)
#    define USE_ASYNC_LDS 1
#  endif
#endif
#ifndef USE_ASYNC_LDS
#  define USE_ASYNC_LDS 0
#endif

constexpr int EPI_BF16 = 0;   // bf16 store, optional bias
constexpr int EPI_F32  = 1;   // f32 store, optional bias
constexpr int EPI_QK   = 2;   // bias + scatter to [B,H,S,hd] bf16
constexpr int EPI_VT   = 3;   // bias + scatter to [B,H,hd,S] bf16 (transposed V)
constexpr int EPI_GELU = 4;   // bias + exact gelu, bf16 store

__device__ __forceinline__ float gelu_exact(float x) {
    return 0.5f * x * (1.0f + erff(x * 0.70710678118654752440f));
}

// C[M,N] = A[M,K] (row-major bf16) * Bt[N,K]^T (row-major bf16) + bias.
// M, N, K exact multiples of TM, TN, BK (true for every launch below):
// no guards, straight-line async tile loads, double-buffered LDS.
// SDIM/HN/HDIM are compile-time (powers of two) so scatter epilogues are
// shift/mask only.
template<int EPI, int TM, int TN, int WROWS, int WCOLS,
         int SDIM = 2048, int HN = 16, int HDIM = 64>
__global__ __launch_bounds__(256)
void gemm_bf16_wmma(const __bf16* __restrict__ A, int lda,
                    const __bf16* __restrict__ Bt, int ldb,
                    void* __restrict__ Cp, int ldc,
                    const float* __restrict__ bias,
                    int M, int N, int K)
{
    constexpr int FI  = TM / WROWS / 16;        // 16x16 frags per wave (M)
    constexpr int FJ  = TN / WCOLS / 16;        // 16x16 frags per wave (N)
    constexpr int CPR = BK / 8;                 // 16B chunks per tile row
    constexpr int ITA = TM * CPR / 256;         // chunks per thread (A tile)
    constexpr int ITB = TN * CPR / 256;         // chunks per thread (B tile)

    __shared__ __bf16 As[2][TM][BK];
    __shared__ __bf16 Bs[2][TN][BK];

    const int tid  = threadIdx.x;
    const int lane = tid & 31;
    const int wave = tid >> 5;
    const int lm   = lane & 15;
    const int half = lane >> 4;
    const int wm   = (wave / WCOLS) * (FI * 16);
    const int wn   = (wave % WCOLS) * (FJ * 16);
    const int m0   = blockIdx.y * TM;
    const int n0   = blockIdx.x * TN;

    f32x8 acc[FI][FJ];
    #pragma unroll
    for (int i = 0; i < FI; ++i)
        #pragma unroll
        for (int j = 0; j < FJ; ++j)
            #pragma unroll
            for (int e = 0; e < 8; ++e) acc[i][j][e] = 0.0f;

#if !USE_ASYNC_LDS
    bf16x8 ra[ITA], rb[ITB];
#endif

    // ---- issue loads of a k-tile into LDS buffer `buf` ----
    auto issue_tile = [&](int buf, int kk) {
        #pragma unroll
        for (int it = 0; it < ITA; ++it) {
            int idx = tid + it * 256;
            int r = idx / CPR;
            int c = (idx % CPR) * 8;
            const __bf16* gp = A + (size_t)(m0 + r) * lda + kk + c;
#if USE_ASYNC_LDS
            __builtin_amdgcn_global_load_async_to_lds_b128(
                (AS1 i32x4*)(void*)gp, (AS3 i32x4*)&As[buf][r][c], 0, 0);
#else
            ra[it] = *(const bf16x8*)gp;
#endif
        }
        #pragma unroll
        for (int it = 0; it < ITB; ++it) {
            int idx = tid + it * 256;
            int r = idx / CPR;
            int c = (idx % CPR) * 8;
            const __bf16* gp = Bt + (size_t)(n0 + r) * ldb + kk + c;
#if USE_ASYNC_LDS
            __builtin_amdgcn_global_load_async_to_lds_b128(
                (AS1 i32x4*)(void*)gp, (AS3 i32x4*)&Bs[buf][r][c], 0, 0);
#else
            rb[it] = *(const bf16x8*)gp;
#endif
        }
    };
#if !USE_ASYNC_LDS
    auto commit_tile = [&](int buf) {
        #pragma unroll
        for (int it = 0; it < ITA; ++it) {
            int idx = tid + it * 256;
            int r = idx / CPR;
            int c = (idx % CPR) * 8;
            *(bf16x8*)&As[buf][r][c] = ra[it];
        }
        #pragma unroll
        for (int it = 0; it < ITB; ++it) {
            int idx = tid + it * 256;
            int r = idx / CPR;
            int c = (idx % CPR) * 8;
            *(bf16x8*)&Bs[buf][r][c] = rb[it];
        }
    };
#endif

    // prologue: fill buffer 0
    issue_tile(0, 0);
#if USE_ASYNC_LDS
    __builtin_amdgcn_s_wait_asynccnt(0);
#else
    commit_tile(0);
#endif
    __syncthreads();

    const int nk = K / BK;
    for (int kt = 0; kt < nk; ++kt) {
        const int cur = kt & 1;
        const int nxt = cur ^ 1;
        const bool has_next = (kt + 1) < nk;
        if (has_next) issue_tile(nxt, (kt + 1) * BK);

        // two WMMA k-steps (k=32 each) out of the BK=64 LDS tile
        #pragma unroll
        for (int kk2 = 0; kk2 < BK / 32; ++kk2) {
            // A fragment (16x32): lane half -> K = 8h..8h+7 and 16+8h..+7
            bf16x16 afrag[FI];
            #pragma unroll
            for (int i = 0; i < FI; ++i) {
                const __bf16* p = &As[cur][wm + i * 16 + lm][kk2 * 32 + half * 8];
                bf16x8 lo = *(const bf16x8*)p;
                bf16x8 hi = *(const bf16x8*)(p + 16);
                afrag[i] = __builtin_shufflevector(lo, hi,
                    0,1,2,3,4,5,6,7,8,9,10,11,12,13,14,15);
            }
            // B fragment (32x16): lane half -> K = 8h..8h+7 and 16+8h..+7
            bf16x16 bfrag[FJ];
            #pragma unroll
            for (int j = 0; j < FJ; ++j) {
                const __bf16* p = &Bs[cur][wn + j * 16 + lm][kk2 * 32 + half * 8];
                bf16x8 lo = *(const bf16x8*)p;
                bf16x8 hi = *(const bf16x8*)(p + 16);
                bfrag[j] = __builtin_shufflevector(lo, hi,
                    0,1,2,3,4,5,6,7,8,9,10,11,12,13,14,15);
            }
            #pragma unroll
            for (int i = 0; i < FI; ++i)
                #pragma unroll
                for (int j = 0; j < FJ; ++j)
                    acc[i][j] = __builtin_amdgcn_wmma_f32_16x16x32_bf16(
                        false, afrag[i], false, bfrag[j],
                        (short)0, acc[i][j], false, false);
        }

        if (has_next) {
#if USE_ASYNC_LDS
            __builtin_amdgcn_s_wait_asynccnt(0);
#else
            commit_tile(nxt);
#endif
            __syncthreads();
        }
    }

    // Epilogue: C layout: vgpr r -> m = r + 8*half, lane%16 -> n
    #pragma unroll
    for (int i = 0; i < FI; ++i) {
        #pragma unroll
        for (int j = 0; j < FJ; ++j) {
            int n = n0 + wn + j * 16 + lm;
            float bv = (bias != nullptr) ? bias[n] : 0.0f;
            #pragma unroll
            for (int r = 0; r < 8; ++r) {
                int m = m0 + wm + i * 16 + r + half * 8;
                float v = acc[i][j][r] + bv;
                if (EPI == EPI_F32) {
                    ((float*)Cp)[(size_t)m * ldc + n] = v;
                } else if (EPI == EPI_BF16) {
                    ((__bf16*)Cp)[(size_t)m * ldc + n] = (__bf16)v;
                } else if (EPI == EPI_GELU) {
                    ((__bf16*)Cp)[(size_t)m * ldc + n] = (__bf16)gelu_exact(v);
                } else if (EPI == EPI_QK) {
                    int b = m / SDIM, s = m % SDIM;   // power-of-two: shifts
                    int h = n / HDIM, d = n % HDIM;
                    ((__bf16*)Cp)[((size_t)(b * HN + h) * SDIM + s) * HDIM + d] = (__bf16)v;
                } else if (EPI == EPI_VT) {
                    int b = m / SDIM, s = m % SDIM;
                    int h = n / HDIM, d = n % HDIM;
                    ((__bf16*)Cp)[((size_t)(b * HN + h) * HDIM + d) * SDIM + s] = (__bf16)v;
                }
            }
        }
    }
}

__global__ __launch_bounds__(256)
void cvt_f32_to_bf16(const float* __restrict__ x, __bf16* __restrict__ y, int n) {
    int i = blockIdx.x * 256 + threadIdx.x;
    if (i < n) y[i] = (__bf16)x[i];
}

// W: K x N row-major (f32) -> WT: N x K row-major (bf16)
__global__ __launch_bounds__(256)
void cvt_transpose_bf16(const float* __restrict__ W, __bf16* __restrict__ WT,
                        int K, int N) {
    long i = (long)blockIdx.x * 256 + threadIdx.x;
    long total = (long)K * N;
    if (i < total) {
        int n = (int)(i / K);
        int k = (int)(i % K);
        WT[i] = (__bf16)W[(size_t)k * N + n];
    }
}

__global__ __launch_bounds__(256)
void softmax_diag_kernel(const float* __restrict__ scores, __bf16* __restrict__ attn,
                         int S, float scale) {
    __shared__ float red[256];
    int row = blockIdx.x;
    int tid = threadIdx.x;
    const float* sr = scores + (size_t)row * S;
    float mx = -3.4e38f;
    for (int c = tid; c < S; c += 256) mx = fmaxf(mx, sr[c] * scale);
    red[tid] = mx; __syncthreads();
    for (int s = 128; s > 0; s >>= 1) {
        if (tid < s) red[tid] = fmaxf(red[tid], red[tid + s]);
        __syncthreads();
    }
    mx = red[0]; __syncthreads();
    float sum = 0.0f;
    for (int c = tid; c < S; c += 256) sum += __expf(sr[c] * scale - mx);
    red[tid] = sum; __syncthreads();
    for (int s = 128; s > 0; s >>= 1) {
        if (tid < s) red[tid] += red[tid + s];
        __syncthreads();
    }
    float inv = 1.0f / red[0];
    for (int c = tid; c < S; c += 256) {
        float e = __expf(sr[c] * scale - mx) * inv;
        if (c == row) e = 0.0f;   // zero diagonal post-softmax
        attn[(size_t)row * S + c] = (__bf16)e;
    }
}

__global__ __launch_bounds__(256)
void layernorm_kernel(const float* __restrict__ h, const float* __restrict__ gamma,
                      const float* __restrict__ beta, float* __restrict__ out, int D) {
    __shared__ float r1[256];
    __shared__ float r2[256];
    int row = blockIdx.x;
    int tid = threadIdx.x;
    const float* hr = h + (size_t)row * D;
    float s = 0.0f, s2 = 0.0f;
    for (int c = tid; c < D; c += 256) { float v = hr[c]; s += v; s2 += v * v; }
    r1[tid] = s; r2[tid] = s2; __syncthreads();
    for (int k = 128; k > 0; k >>= 1) {
        if (tid < k) { r1[tid] += r1[tid + k]; r2[tid] += r2[tid + k]; }
        __syncthreads();
    }
    float mu  = r1[0] / D;
    float var = r2[0] / D - mu * mu;
    float inv = rsqrtf(var + 1e-5f);
    for (int c = tid; c < D; c += 256)
        out[(size_t)row * D + c] = (hr[c] - mu) * inv * gamma[c] + beta[c];
}

extern "C" void kernel_launch(void* const* d_in, const int* in_sizes, int n_in,
                              void* d_out, int out_size, void* d_ws, size_t ws_size,
                              hipStream_t stream) {
    const float* x   = (const float*)d_in[0];
    const float* Wq  = (const float*)d_in[1];
    const float* bq  = (const float*)d_in[2];
    const float* Wk  = (const float*)d_in[3];
    const float* bk  = (const float*)d_in[4];
    const float* Wv  = (const float*)d_in[5];
    const float* bv  = (const float*)d_in[6];
    const float* W1  = (const float*)d_in[7];
    const float* b1  = (const float*)d_in[8];
    const float* W2  = (const float*)d_in[9];
    const float* b2  = (const float*)d_in[10];
    const float* gamma = (const float*)d_in[11];
    const float* beta  = (const float*)d_in[12];
    float* out = (float*)d_out;

    const int B = 2, S = 2048, D = 1024, H = 16, HD = 64, F = 4096;
    const int M = B * S;              // 4096 token rows
    const float scale = 1.0f / 32.0f; // 1/sqrt(D)

    char* ws = (char*)d_ws;
    size_t off = 0;
    auto alloc = [&](size_t bytes) -> void* {
        void* p = ws + off;
        off += (bytes + 255) & ~((size_t)255);
        return p;
    };
    __bf16* xb   = (__bf16*)alloc((size_t)M * D * 2);           // x bf16
    __bf16* wqT  = (__bf16*)alloc((size_t)D * D * 2);           // Wq^T (N,K)
    __bf16* wkT  = (__bf16*)alloc((size_t)D * D * 2);
    __bf16* wvT  = (__bf16*)alloc((size_t)D * D * 2);
    __bf16* w1T  = (__bf16*)alloc((size_t)D * F * 2);           // [F, D]
    __bf16* w2T  = (__bf16*)alloc((size_t)F * D * 2);           // [D, F]
    __bf16* Qh   = (__bf16*)alloc((size_t)B * H * S * HD * 2);  // [B,H,S,hd]
    __bf16* Kh   = (__bf16*)alloc((size_t)B * H * S * HD * 2);  // [B,H,S,hd]
    __bf16* Vt   = (__bf16*)alloc((size_t)B * H * HD * S * 2);  // [B,H,hd,S]
    __bf16* aout = (__bf16*)alloc((size_t)M * D * 2);           // attention out bf16
    __bf16* h1   = (__bf16*)alloc((size_t)M * F * 2);           // gelu(ffn1) bf16
    float*  hbuf = (float*) alloc((size_t)M * D * 4);           // pre-LN f32
    float*  scb  = (float*) alloc((size_t)S * S * 4);           // per-head scores f32
    __bf16* atb  = (__bf16*)alloc((size_t)S * S * 2);           // per-head attn bf16

    // --- conversions ---
    {
        int n = M * D;
        cvt_f32_to_bf16<<<(n + 255) / 256, 256, 0, stream>>>(x, xb, n);
    }
    {
        long n = (long)D * D;
        dim3 g((unsigned)((n + 255) / 256));
        cvt_transpose_bf16<<<g, 256, 0, stream>>>(Wq, wqT, D, D);
        cvt_transpose_bf16<<<g, 256, 0, stream>>>(Wk, wkT, D, D);
        cvt_transpose_bf16<<<g, 256, 0, stream>>>(Wv, wvT, D, D);
    }
    {
        long n = (long)D * F;
        dim3 g((unsigned)((n + 255) / 256));
        cvt_transpose_bf16<<<g, 256, 0, stream>>>(W1, w1T, D, F);
        cvt_transpose_bf16<<<g, 256, 0, stream>>>(W2, w2T, F, D);
    }

    // --- QKV projections (fused head scatter) ---
    {
        dim3 g(D / 128, M / 128);
        gemm_bf16_wmma<EPI_QK, 128, 128, 2, 4><<<g, 256, 0, stream>>>(
            xb, D, wqT, D, (void*)Qh, D, bq, M, D, D);
        gemm_bf16_wmma<EPI_QK, 128, 128, 2, 4><<<g, 256, 0, stream>>>(
            xb, D, wkT, D, (void*)Kh, D, bk, M, D, D);
        gemm_bf16_wmma<EPI_VT, 128, 128, 2, 4><<<g, 256, 0, stream>>>(
            xb, D, wvT, D, (void*)Vt, D, bv, M, D, D);
    }

    // --- attention, per (b,h), reusing score/attn scratch ---
    for (int bh = 0; bh < B * H; ++bh) {
        int b = bh / H;
        int h = bh % H;
        const __bf16* Qp = Qh + (size_t)bh * S * HD;
        const __bf16* Kp = Kh + (size_t)bh * S * HD;
        const __bf16* Vp = Vt + (size_t)bh * HD * S;
        // scores = Q * K^T  (S x S, K-dim = hd = one LDS tile)
        {
            dim3 g(S / 128, S / 128);
            gemm_bf16_wmma<EPI_F32, 128, 128, 2, 4><<<g, 256, 0, stream>>>(
                Qp, HD, Kp, HD, (void*)scb, S, nullptr, S, S, HD);
        }
        // softmax (scaled) + zero diagonal -> bf16
        softmax_diag_kernel<<<S, 256, 0, stream>>>(scb, atb, S, scale);
        // out = attn * V  (S x hd, K-dim = S); write into [b, s, h*hd + d]
        {
            __bf16* Cptr = aout + (size_t)b * S * D + (size_t)h * HD;
            dim3 g(HD / 64, S / 128);
            gemm_bf16_wmma<EPI_BF16, 128, 64, 4, 2><<<g, 256, 0, stream>>>(
                atb, S, Vp, S, (void*)Cptr, D, nullptr, S, HD, S);
        }
    }

    // --- FFN ---
    {
        dim3 g(F / 128, M / 128);
        gemm_bf16_wmma<EPI_GELU, 128, 128, 2, 4><<<g, 256, 0, stream>>>(
            aout, D, w1T, D, (void*)h1, F, b1, M, F, D);
    }
    {
        dim3 g(D / 128, M / 128);
        gemm_bf16_wmma<EPI_F32, 128, 128, 2, 4><<<g, 256, 0, stream>>>(
            h1, F, w2T, F, (void*)hbuf, D, b2, M, D, F);
    }

    // --- LayerNorm ---
    layernorm_kernel<<<M, 256, 0, stream>>>(hbuf, gamma, beta, out, D);
}
